// replacement_class_80487687127373
// MI455X (gfx1250) — 8-bit matrix operand order repaired and checked by reading, compiled, not run on hardware
//
#include <hip/hip_runtime.h>

typedef int v8i __attribute__((ext_vector_type(8)));
typedef int v4i __attribute__((ext_vector_type(4)));

#define TILE_M 128
#define TILE_N 128
#define TILE_K 64
#define LDS_STRIDE 80
#define BUF_BYTES (TILE_M * LDS_STRIDE)

__global__ __launch_bounds__(256) void quant_x_kernel(
    const float* __restrict__ x, const float* __restrict__ x_scale,
    int* __restrict__ xq, int n4) {
  int i = blockIdx.x * 256 + threadIdx.x;
  if (i >= n4) return;
  const float inv = 1.0f / x_scale[0];
  float4 v = ((const float4*)x)[i];
  int b0 = (int)fminf(fmaxf(rintf(v.x * inv), -128.0f), 127.0f);
  int b1 = (int)fminf(fmaxf(rintf(v.y * inv), -128.0f), 127.0f);
  int b2 = (int)fminf(fmaxf(rintf(v.z * inv), -128.0f), 127.0f);
  int b3 = (int)fminf(fmaxf(rintf(v.w * inv), -128.0f), 127.0f);
  xq[i] = (b0 & 255) | ((b1 & 255) << 8) | ((b2 & 255) << 16) | ((b3 & 255) << 24);
}

__global__ __launch_bounds__(256) void pack_wt_kernel(
    const int* __restrict__ w, signed char* __restrict__ wt, int K, int N) {
  __shared__ signed char tile[64 * 68];
  const int n0 = blockIdx.x * 64;
  const int k0 = blockIdx.y * 64;
  const int t = threadIdx.x;
#pragma unroll
  for (int i = 0; i < 16; ++i) {
    int linear = i * 256 + t;
    int kk = linear >> 6;
    int nn = linear & 63;
    int val = w[(size_t)(k0 + kk) * N + (n0 + nn)];
    tile[nn * 68 + kk] = (signed char)val;
  }
  __syncthreads();
  const int nn = t >> 2;
  const int q = t & 3;
  v4i d;
#pragma unroll
  for (int j = 0; j < 4; ++j)
    d[j] = *(const int*)&tile[nn * 68 + q * 16 + j * 4];
  *(v4i*)&wt[(size_t)(n0 + nn) * K + k0 + q * 16] = d;
}

__global__ __launch_bounds__(256) void gemm_i8_kernel(
    const signed char* __restrict__ xq, const signed char* __restrict__ wt,
    const float* __restrict__ x_scale, const float* __restrict__ w_scale,
    float* __restrict__ out, int M, int N, int K) {
  __shared__ signed char As[2][BUF_BYTES];
  __shared__ signed char Bs[2][BUF_BYTES];

  const int tid = threadIdx.x;
  const int lane = tid & 31;
  const int l15 = lane & 15;
  const int h = (lane >> 4) & 1;
  const int wid = tid >> 5;
  const int wm = wid & 3;
  const int wn = wid >> 2;
  const int m0 = blockIdx.y * TILE_M;
  const int n0 = blockIdx.x * TILE_N;

  const int r0 = tid >> 2;
  const int r1 = (tid >> 2) + 64;
  const int c0 = tid & 3;

  const unsigned voff0 = (unsigned)(r0 * K + c0 * 16);
  const unsigned voff1 = (unsigned)(r1 * K + c0 * 16);
  const unsigned ldsOff0 = (unsigned)(r0 * LDS_STRIDE + c0 * 16);
  const unsigned ldsOff1 = (unsigned)(r1 * LDS_STRIDE + c0 * 16);
  const unsigned aBase = (unsigned)(size_t)&As[0][0];
  const unsigned bBase = (unsigned)(size_t)&Bs[0][0];
  const unsigned long long gA = (unsigned long long)(xq + (size_t)m0 * K);
  const unsigned long long gB = (unsigned long long)(wt + (size_t)n0 * K);

  auto issue_tile = [&](int buf, int kt) {
    const unsigned ab = aBase + (unsigned)(buf * BUF_BYTES);
    const unsigned bb = bBase + (unsigned)(buf * BUF_BYTES);
    const unsigned long long ga = gA + (unsigned)kt;
    const unsigned long long gb = gB + (unsigned)kt;
    asm volatile(
        "global_load_async_to_lds_b128 %0, %4, %6 offset:0\n\t"
        "global_load_async_to_lds_b128 %1, %5, %6 offset:0\n\t"
        "global_load_async_to_lds_b128 %2, %4, %7 offset:0\n\t"
        "global_load_async_to_lds_b128 %3, %5, %7 offset:0"
        :
        : "v"(ab + ldsOff0), "v"(ab + ldsOff1),
          "v"(bb + ldsOff0), "v"(bb + ldsOff1),
          "v"(voff0), "v"(voff1), "s"(ga), "s"(gb)
        : "memory");
  };
#define WAIT_ASYNC() asm volatile("s_wait_asynccnt 0x0" ::: "memory")

  v8i acc[2][4];
#pragma unroll
  for (int i = 0; i < 2; ++i)
#pragma unroll
    for (int j = 0; j < 4; ++j) acc[i][j] = (v8i)0;

  issue_tile(0, 0);
  WAIT_ASYNC();
  __syncthreads();

  for (int kt = 0; kt < K; kt += TILE_K) {
    const int cur = (kt / TILE_K) & 1;
    const bool more = (kt + TILE_K) < K;
    if (more) issue_tile(cur ^ 1, kt + TILE_K);
    if (kt + 2 * TILE_K < K) {
      __builtin_prefetch(&xq[(size_t)(m0 + r0) * K + kt + 2 * TILE_K + c0 * 16], 0, 3);
      __builtin_prefetch(&wt[(size_t)(n0 + r0) * K + kt + 2 * TILE_K + c0 * 16], 0, 3);
    }

    v8i Af[2], Bf[4];
#pragma unroll
    for (int tm = 0; tm < 2; ++tm) {
      const int base = (wm * 32 + tm * 16 + l15) * LDS_STRIDE + h * 8;
#pragma unroll
      for (int v = 0; v < 8; ++v)
        Af[tm][v] = *(const int*)&As[cur][base + ((v >> 1) * 16) + ((v & 1) * 4)];
    }
#pragma unroll
    for (int tn = 0; tn < 4; ++tn) {
      const int base = (wn * 64 + tn * 16 + l15) * LDS_STRIDE + h * 8;
#pragma unroll
      for (int v = 0; v < 8; ++v)
        Bf[tn][v] = *(const int*)&Bs[cur][base + ((v >> 1) * 16) + ((v & 1) * 4)];
    }

#pragma unroll
    for (int tm = 0; tm < 2; ++tm)
#pragma unroll
      for (int tn = 0; tn < 4; ++tn)
        acc[tm][tn] = __builtin_amdgcn_wmma_i32_16x16x64_iu8(
             true, Af[tm],  true, Bf[tn], acc[tm][tn],
             false,  false);

    if (more) {
      WAIT_ASYNC();
      __syncthreads();
    }
  }

  const float xs = x_scale[0];
#pragma unroll
  for (int tn = 0; tn < 4; ++tn) {
    const int n = n0 + wn * 64 + tn * 16 + l15;
    const float sc = xs * w_scale[n];
#pragma unroll
    for (int tm = 0; tm < 2; ++tm) {
      const int mb = m0 + wm * 32 + tm * 16 + h * 8;
#pragma unroll
      for (int r = 0; r < 8; ++r)
        out[(size_t)(mb + r) * N + n] = (float)acc[tm][tn][r] * sc;
    }
  }
}

extern "C" void kernel_launch(void* const* d_in, const int* in_sizes, int n_in,
                              void* d_out, int out_size, void* d_ws, size_t ws_size,
                              hipStream_t stream) {
  const float* x = (const float*)d_in[0];
  const float* x_scale = (const float*)d_in[1];
  const int* w = (const int*)d_in[2];
  const float* w_scale = (const float*)d_in[3];
  float* out = (float*)d_out;

  const int N = in_sizes[3];
  const int K = in_sizes[2] / N;
  const int M = in_sizes[0] / K;

  int* xq = (int*)d_ws;
  signed char* wt = (signed char*)d_ws + (size_t)M * K;

  const int n4 = (M * K) / 4;
  quant_x_kernel<<<(n4 + 255) / 256, 256, 0, stream>>>(x, x_scale, xq, n4);

  dim3 pgrid(N / 64, K / 64);
  pack_wt_kernel<<<pgrid, 256, 0, stream>>>(w, wt, K, N);

  dim3 ggrid(N / TILE_N, M / TILE_M);
  gemm_i8_kernel<<<ggrid, 256, 0, stream>>>((const signed char*)xq, wt,
                                            x_scale, w_scale, out, M, N, K);
}
